// LVAwareDiffPool_22239340659050
// MI455X (gfx1250) — matrix operand order repaired and checked by reading, not compiled, not run on hardware
//
#include <hip/hip_runtime.h>

#define N_BUILD 200000
#define N_FEAT  128
#define HID     64
#define N_GROUPS 8192
#define MIN_SZ  3
#define MAX_SZ  20
#define TC      (N_GROUPS * MAX_SZ)

typedef __bf16 v16bf __attribute__((ext_vector_type(16)));
typedef float  v8f   __attribute__((ext_vector_type(8)));

// ---------------------------------------------------------------------------
// WMMA helpers (wave32, CDNA5 ISA 7.12.2 layouts, fragment-major storage)
// ---------------------------------------------------------------------------
__device__ __forceinline__ v8f wmma_bf16(v16bf a, v16bf b, v8f c) {
  return __builtin_amdgcn_wmma_f32_16x16x32_bf16(
      false, a, false, b, (short)0, c, false, false);
}

// A fragment element mapping (16x32 bf16): lane = half*16 + m,
//   element e <-> k = ((e>>3)<<4) + half*8 + (e&7)
// B fragment element mapping (32x16 bf16): lane = half*16 + n,
//   element e <-> k = ((e>>3)<<4) + half*8 + (e&7)
// Fragments are stored "fragment-major": 32B per lane, contiguous.

// swizzled A load from LDS: one aligned 32-byte read per lane
__device__ __forceinline__ v16bf load_a_sw(const __bf16* __restrict__ A,
                                           int ks, int lane) {
  return *(const v16bf*)(A + ((size_t)(ks * 32 + lane) << 4));
}

// swizzled B load from global: one aligned 32-byte read per lane
__device__ __forceinline__ v16bf load_b_sw(const __bf16* __restrict__ B,
                                           int nnt, int ks, int nt, int lane) {
  return *(const v16bf*)(B + ((size_t)((ks * nnt + nt) * 32 + lane) << 4));
}

// scatter one f32 value (row m, k-index k) into swizzled bf16 A storage
__device__ __forceinline__ void store_a_sw(__bf16* __restrict__ A,
                                           int m, int k, float v) {
  int ks = k >> 5, kk = k & 31;
  int half = (kk >> 3) & 1;
  int e = ((kk >> 4) << 3) + (kk & 7);
  A[(((ks * 32) + half * 16 + m) << 4) + e] = (__bf16)v;
}

// A fragment for layer 1 built directly from global row-major f32 x:
// two contiguous 8-float runs per lane -> 4x float4 loads, no LDS.
__device__ __forceinline__ v16bf load_a_x(const float* __restrict__ xr,
                                          int k0, int lane) {
  const int half = (lane >> 4) & 1;
  const float4* p0 = (const float4*)(xr + k0 + half * 8);
  const float4* p1 = (const float4*)(xr + k0 + 16 + half * 8);
  float4 a0 = p0[0], a1 = p0[1];
  float4 b0 = p1[0], b1 = p1[1];
  v16bf a;
  a[0] = (__bf16)a0.x;  a[1] = (__bf16)a0.y;  a[2] = (__bf16)a0.z;  a[3] = (__bf16)a0.w;
  a[4] = (__bf16)a1.x;  a[5] = (__bf16)a1.y;  a[6] = (__bf16)a1.z;  a[7] = (__bf16)a1.w;
  a[8] = (__bf16)b0.x;  a[9] = (__bf16)b0.y;  a[10] = (__bf16)b0.z; a[11] = (__bf16)b0.w;
  a[12] = (__bf16)b1.x; a[13] = (__bf16)b1.y; a[14] = (__bf16)b1.z; a[15] = (__bf16)b1.w;
  return a;
}

// ---------------------------------------------------------------------------
// Kernel 1: convert weights f32 -> bf16 in fragment-major (swizzled) layout
// ---------------------------------------------------------------------------
__device__ __forceinline__ void swizzle_b(const float* __restrict__ w,
                                          __bf16* __restrict__ out,
                                          int N, int Nsrc, int i) {
  // i indexes the swizzled output [K*N)
  int e = i & 15;
  int lane = (i >> 4) & 31;
  int tile = i >> 9;
  int nnt = N >> 4;
  int nt = tile % nnt, ks = tile / nnt;
  int half = (lane >> 4) & 1, n = lane & 15;
  int k = ks * 32 + ((e >> 3) << 4) + half * 8 + (e & 7);
  int col = nt * 16 + n;
  float v = (col < Nsrc) ? w[k * Nsrc + col] : 0.0f;
  out[i] = (__bf16)v;
}

__global__ void cvt_weights(const float* __restrict__ wa1,
                            const float* __restrict__ wa2,
                            const float* __restrict__ wa3,
                            const float* __restrict__ wq1,
                            __bf16* __restrict__ o1, __bf16* __restrict__ o2,
                            __bf16* __restrict__ o3, __bf16* __restrict__ oq1) {
  int i = blockIdx.x * blockDim.x + threadIdx.x;
  if (i < N_FEAT * HID)     swizzle_b(wa1, o1, HID, HID, i);       // 128x64
  if (i < HID * HID)        swizzle_b(wa2, o2, HID, HID, i);       // 64x64
  if (i < HID * 32)         swizzle_b(wa3, o3, 32, MAX_SZ, i);     // 64x32 pad
  if (i < 2 * N_FEAT * HID) swizzle_b(wq1, oq1, HID, HID, i);      // 256x64
}

// ---------------------------------------------------------------------------
// Kernel 2: per-LV-group counts + feature sums (atomic segment sum)
// ---------------------------------------------------------------------------
__global__ void group_accum(const float* __restrict__ x,
                            const int* __restrict__ gid,
                            float* __restrict__ nb, float* __restrict__ gsum) {
  int i = blockIdx.x * blockDim.x + threadIdx.x;
  if (i >= N_BUILD * N_FEAT) return;
  int n = i >> 7, f = i & 127;
  int g = gid[n];
  atomicAdd(&gsum[g * N_FEAT + f], x[i]);
  if (f == 0) atomicAdd(&nb[g], 1.0f);
}

// ---------------------------------------------------------------------------
// Kernel 3: group mean + cluster-count MLP -> k[g]  (tiny; scalar VALU)
// ---------------------------------------------------------------------------
__global__ void cc_mlp(const float* __restrict__ nb, const float* __restrict__ gsum,
                       const float* __restrict__ w1, const float* __restrict__ b1,
                       const float* __restrict__ w2, const float* __restrict__ b2,
                       const float* __restrict__ w3, const float* __restrict__ b3,
                       float* __restrict__ gmean, int* __restrict__ kG) {
  int g = blockIdx.x * blockDim.x + threadIdx.x;
  if (g >= N_GROUPS) return;
  float n = nb[g];
  float inv = 1.0f / fmaxf(n, 1.0f);
  float h1[HID];
#pragma unroll
  for (int j = 0; j < HID; ++j) h1[j] = b1[j];
  for (int f = 0; f < N_FEAT; ++f) {
    float xv = gsum[g * N_FEAT + f] * inv;
    gmean[g * N_FEAT + f] = xv;
#pragma unroll
    for (int j = 0; j < HID; ++j) h1[j] += xv * w1[f * HID + j];
  }
  float h2[HID / 2];
#pragma unroll
  for (int j = 0; j < HID / 2; ++j) h2[j] = b2[j];
  for (int f = 0; f < HID; ++f) {
    float v = fmaxf(h1[f], 0.0f);
#pragma unroll
    for (int j = 0; j < HID / 2; ++j) h2[j] += v * w2[f * (HID / 2) + j];
  }
  float s = b3[0];
#pragma unroll
  for (int j = 0; j < HID / 2; ++j) s += fmaxf(h2[j], 0.0f) * w3[j];
  float ratio = 1.0f / (1.0f + __expf(-s));
  int ni = (int)n;
  int maxp = min(ni / MIN_SZ, MAX_SZ);
  int minp = max(1, ni / MAX_SZ);
  int k = minp + (int)floorf(ratio * (float)(maxp - minp));
  k = min(max(k, minp), maxp);
  if (ni < MIN_SZ) k = 0;
  kG[g] = k;
}

// ---------------------------------------------------------------------------
// Kernel 4: assignment MLP (WMMA) + masked softmax + size accumulation
// One wave = one 16-row tile; 4 waves per block.
// ---------------------------------------------------------------------------
#define AMW 4
__global__ void __launch_bounds__(128) assign_mlp_kernel(
    const float* __restrict__ x, const int* __restrict__ gid,
    const int* __restrict__ kG,
    const __bf16* __restrict__ w1, const float* __restrict__ b1,
    const __bf16* __restrict__ w2, const float* __restrict__ b2,
    const __bf16* __restrict__ w3, const float* __restrict__ b3,
    float* __restrict__ soft_out, float* __restrict__ sizes) {
  __shared__ __attribute__((aligned(32))) __bf16 stA[AMW][2 * 32 * 16]; // 2KB/wave
  __shared__ float stC[AMW][16 * 32];                                   // 2KB/wave
  const int lane = threadIdx.x & 31;
  const int wv = threadIdx.x >> 5;
  const int half = (lane >> 4) & 1, nn = lane & 15, m16 = lane & 15;
  __bf16* A = stA[wv];
  float*  C = stC[wv];
  const int NT = N_BUILD / 16;               // 12500, exact
  for (int t = blockIdx.x * AMW + wv; t < NT; t += gridDim.x * AMW) {
    const int row0 = t * 16;
    const float* xr = x + (size_t)(row0 + m16) * N_FEAT;

    // ---- layer 1: [16,128] @ [128,64], A straight from global ----
    v16bf af[4];
#pragma unroll
    for (int ks = 0; ks < 4; ++ks) af[ks] = load_a_x(xr, ks * 32, lane);
    v8f c1[4];
#pragma unroll
    for (int nt = 0; nt < 4; ++nt) {
      v8f acc = {};
#pragma unroll
      for (int ks = 0; ks < 4; ++ks)
        acc = wmma_bf16(af[ks], load_b_sw(w1, 4, ks, nt, lane), acc);
      c1[nt] = acc;
    }
    // relu+bias -> swizzled bf16 A staging (k index = h1 column)
#pragma unroll
    for (int nt = 0; nt < 4; ++nt) {
      int nh = nt * 16 + nn;
      float bias = b1[nh];
#pragma unroll
      for (int r = 0; r < 8; ++r)
        store_a_sw(A, half * 8 + r, nh, fmaxf(c1[nt][r] + bias, 0.0f));
    }
    asm volatile("s_wait_dscnt 0" ::: "memory");

    // ---- layer 2: [16,64] @ [64,64] ----
    v16bf a2[2];
#pragma unroll
    for (int ks = 0; ks < 2; ++ks) a2[ks] = load_a_sw(A, ks, lane);
    v8f c2[4];
#pragma unroll
    for (int nt = 0; nt < 4; ++nt) {
      v8f acc = {};
#pragma unroll
      for (int ks = 0; ks < 2; ++ks)
        acc = wmma_bf16(a2[ks], load_b_sw(w2, 4, ks, nt, lane), acc);
      c2[nt] = acc;
    }
#pragma unroll
    for (int nt = 0; nt < 4; ++nt) {
      int nh = nt * 16 + nn;
      float bias = b2[nh];
#pragma unroll
      for (int r = 0; r < 8; ++r)
        store_a_sw(A, half * 8 + r, nh, fmaxf(c2[nt][r] + bias, 0.0f));
    }
    asm volatile("s_wait_dscnt 0" ::: "memory");

    // ---- layer 3: [16,64] @ [64,32] (cols 20..31 zero-padded) ----
    v16bf a3[2];
#pragma unroll
    for (int ks = 0; ks < 2; ++ks) a3[ks] = load_a_sw(A, ks, lane);
#pragma unroll
    for (int nt = 0; nt < 2; ++nt) {
      v8f acc = {};
#pragma unroll
      for (int ks = 0; ks < 2; ++ks)
        acc = wmma_bf16(a3[ks], load_b_sw(w3, 2, ks, nt, lane), acc);
      int col = nt * 16 + nn;
      float bias = (col < MAX_SZ) ? b3[col] : 0.0f;
#pragma unroll
      for (int r = 0; r < 8; ++r)
        C[(half * 8 + r) * 32 + col] = acc[r] + bias;
    }
    asm volatile("s_wait_dscnt 0" ::: "memory");

    // ---- masked softmax per row (lanes 0..15, one row each) ----
    if (lane < 16) {
      int row = lane;
      int g = gid[row0 + row];
      int kv = kG[g];
      float mx = -1e30f;
      for (int s = 0; s < kv; ++s) mx = fmaxf(mx, C[row * 32 + s]);
      float ev[MAX_SZ];
      float sum = 0.0f;
      for (int s = 0; s < kv; ++s) {
        float e = __expf(C[row * 32 + s] - mx);
        ev[s] = e;
        sum += e;
      }
      float invs = (kv > 0) ? 1.0f / sum : 0.0f;
      for (int s = 0; s < MAX_SZ; ++s) {
        float v = (s < kv) ? ev[s] * invs : 0.0f;
        soft_out[(row0 + row) * MAX_SZ + s] = v;
        if (v != 0.0f) atomicAdd(&sizes[g * MAX_SZ + s], v);
      }
    }
  }
}

// ---------------------------------------------------------------------------
// Kernel 5: min-size correction, corrected soft, argmax -> cid, cluster count
// ---------------------------------------------------------------------------
__global__ void correct_argmax(const int* __restrict__ gid,
                               const float* __restrict__ sizes,
                               float* __restrict__ soft,        // in/out
                               float* __restrict__ cid_f,
                               int* __restrict__ cid_i,
                               float* __restrict__ cnt) {
  int n = blockIdx.x * blockDim.x + threadIdx.x;
  if (n >= N_BUILD) return;
  int g = gid[n];
  float best = -1e38f;
  int bi = 0;
#pragma unroll
  for (int s = 0; s < MAX_SZ; ++s) {
    float fac = (sizes[g * MAX_SZ + s] < (float)MIN_SZ) ? 0.1f : 1.0f;
    float v = soft[n * MAX_SZ + s] * fac;
    soft[n * MAX_SZ + s] = v;
    if (v > best) { best = v; bi = s; }   // strict > == jnp.argmax first-max
  }
  int c = g * MAX_SZ + bi;
  cid_i[n] = c;
  cid_f[n] = (float)c;
  atomicAdd(&cnt[c], 1.0f);
}

// ---------------------------------------------------------------------------
// Kernel 6: per-hard-cluster feature sums and sums of squares
// ---------------------------------------------------------------------------
__global__ void cluster_accum(const float* __restrict__ x,
                              const int* __restrict__ cid,
                              float* __restrict__ csum, float* __restrict__ csq) {
  int i = blockIdx.x * blockDim.x + threadIdx.x;
  if (i >= N_BUILD * N_FEAT) return;
  int n = i >> 7, f = i & 127;
  int c = cid[n];
  float v = x[i];
  atomicAdd(&csum[c * N_FEAT + f], v);
  atomicAdd(&csq[c * N_FEAT + f], v * v);
}

// ---------------------------------------------------------------------------
// Kernel 7: cluster features + quality MLP (WMMA 256->64, scalar 64->1)
// One wave = 16 clusters; q_in staged swizzled in LDS.
// ---------------------------------------------------------------------------
#define QMW 4
__global__ void __launch_bounds__(128) quality_kernel(
    const float* __restrict__ cnt, const float* __restrict__ csum,
    const float* __restrict__ csq, const float* __restrict__ gmean,
    const __bf16* __restrict__ wq1, const float* __restrict__ bq1,
    const float* __restrict__ wq2, const float* __restrict__ bq2,
    float* __restrict__ cfeat_out, float* __restrict__ quality_out) {
  __shared__ __attribute__((aligned(32))) __bf16 stA[QMW][8 * 32 * 16]; // 8KB/wave
  __shared__ float stH[QMW][16 * HID];                                  // 4KB/wave
  const int lane = threadIdx.x & 31;
  const int wv = threadIdx.x >> 5;
  const int half = (lane >> 4) & 1, nn = lane & 15;
  __bf16* A = stA[wv];
  float*  H = stH[wv];
  const int NT = TC / 16;                    // 10240, exact
  for (int t = blockIdx.x * QMW + wv; t < NT; t += gridDim.x * QMW) {
    const int c0 = t * 16;
    // build q_in = [cmean | std] directly in swizzled bf16 layout
    for (int i = lane; i < 16 * N_FEAT; i += 32) {
      int r = i >> 7, f = i & 127;
      int c = c0 + r;
      float n = cnt[c];
      float mu = csum[c * N_FEAT + f] / fmaxf(n, 1.0f);
      float sq = csq[c * N_FEAT + f];
      float var = (sq - n * mu * mu) / fmaxf(n - 1.0f, 1.0f);
      float sd = sqrtf(fmaxf(var, 0.0f));
      float cf = (n > 0.0f) ? mu : gmean[(c / MAX_SZ) * N_FEAT + f];
      cfeat_out[c * N_FEAT + f] = cf;
      store_a_sw(A, r, f, mu);
      store_a_sw(A, r, N_FEAT + f, sd);
    }
    asm volatile("s_wait_dscnt 0" ::: "memory");

    // [16,256] @ [256,64]
#pragma unroll
    for (int nt = 0; nt < 4; ++nt) {
      v8f acc = {};
#pragma unroll
      for (int ks = 0; ks < 8; ++ks)
        acc = wmma_bf16(load_a_sw(A, ks, lane),
                        load_b_sw(wq1, 4, ks, nt, lane), acc);
      float bias = bq1[nt * 16 + nn];
#pragma unroll
      for (int r = 0; r < 8; ++r)
        H[(half * 8 + r) * HID + nt * 16 + nn] = fmaxf(acc[r] + bias, 0.0f);
    }
    asm volatile("s_wait_dscnt 0" ::: "memory");

    if (lane < 16) {
      int row = lane;
      int c = c0 + row;
      float s = bq2[0];
      for (int j = 0; j < HID; ++j) s += H[row * HID + j] * wq2[j];
      float q = 1.0f / (1.0f + __expf(-s));
      quality_out[c] = (cnt[c] >= 2.0f) ? q : 0.0f;
    }
  }
}

// ---------------------------------------------------------------------------
// Host launcher
// ---------------------------------------------------------------------------
extern "C" void kernel_launch(void* const* d_in, const int* in_sizes, int n_in,
                              void* d_out, int out_size, void* d_ws, size_t ws_size,
                              hipStream_t stream) {
  const float* x     = (const float*)d_in[0];
  // d_in[1] = edge_index: accepted but unused by the reference
  const int*   gid   = (const int*)d_in[2];
  const float* w_cc1 = (const float*)d_in[3];
  const float* b_cc1 = (const float*)d_in[4];
  const float* w_cc2 = (const float*)d_in[5];
  const float* b_cc2 = (const float*)d_in[6];
  const float* w_cc3 = (const float*)d_in[7];
  const float* b_cc3 = (const float*)d_in[8];
  const float* w_a1  = (const float*)d_in[9];
  const float* b_a1  = (const float*)d_in[10];
  const float* w_a2  = (const float*)d_in[11];
  const float* b_a2  = (const float*)d_in[12];
  const float* w_a3  = (const float*)d_in[13];
  const float* b_a3  = (const float*)d_in[14];
  const float* w_q1  = (const float*)d_in[15];
  const float* b_q1  = (const float*)d_in[16];
  const float* w_q2  = (const float*)d_in[17];
  const float* b_q2  = (const float*)d_in[18];

  char* ws = (char*)d_ws;
  size_t off = 0;
  auto alloc = [&](size_t bytes) -> void* {
    void* p = ws + off;
    off += (bytes + 255) & ~(size_t)255;
    return p;
  };
  float* nb    = (float*)alloc((size_t)N_GROUPS * 4);
  float* gsum  = (float*)alloc((size_t)N_GROUPS * N_FEAT * 4);
  float* sizes = (float*)alloc((size_t)N_GROUPS * MAX_SZ * 4);
  float* cnt   = (float*)alloc((size_t)TC * 4);
  float* csum  = (float*)alloc((size_t)TC * N_FEAT * 4);
  float* csq   = (float*)alloc((size_t)TC * N_FEAT * 4);
  size_t zero_bytes = off;
  float* gmean = (float*)alloc((size_t)N_GROUPS * N_FEAT * 4);
  int*   kG    = (int*)alloc((size_t)N_GROUPS * 4);
  int*   cid   = (int*)alloc((size_t)N_BUILD * 4);
  __bf16* w1b  = (__bf16*)alloc((size_t)N_FEAT * HID * 2);
  __bf16* w2b  = (__bf16*)alloc((size_t)HID * HID * 2);
  __bf16* w3b  = (__bf16*)alloc((size_t)HID * 32 * 2);
  __bf16* wq1b = (__bf16*)alloc((size_t)2 * N_FEAT * HID * 2);

  hipMemsetAsync(d_ws, 0, zero_bytes, stream);

  // output layout: soft [N,20] | cid [N] | cfeat [TC,128] | quality [TC]
  float* out     = (float*)d_out;
  float* soft    = out;
  float* cid_f   = soft + (size_t)N_BUILD * MAX_SZ;
  float* cfeat   = cid_f + (size_t)N_BUILD;
  float* quality = cfeat + (size_t)TC * N_FEAT;

  cvt_weights<<<64, 256, 0, stream>>>(w_a1, w_a2, w_a3, w_q1, w1b, w2b, w3b, wq1b);

  group_accum<<<(N_BUILD * N_FEAT) / 256, 256, 0, stream>>>(x, gid, nb, gsum);

  cc_mlp<<<N_GROUPS / 128, 128, 0, stream>>>(nb, gsum, w_cc1, b_cc1, w_cc2, b_cc2,
                                             w_cc3, b_cc3, gmean, kG);

  assign_mlp_kernel<<<512, 128, 0, stream>>>(x, gid, kG, w1b, b_a1, w2b, b_a2,
                                             w3b, b_a3, soft, sizes);

  correct_argmax<<<(N_BUILD + 255) / 256, 256, 0, stream>>>(gid, sizes, soft,
                                                            cid_f, cid, cnt);

  cluster_accum<<<(N_BUILD * N_FEAT) / 256, 256, 0, stream>>>(x, cid, csum, csq);

  quality_kernel<<<640, 128, 0, stream>>>(cnt, csum, csq, gmean, wq1b, b_q1,
                                          w_q2, b_q2, cfeat, quality);
}
